// FeatureEncoder_42451456754085
// MI455X (gfx1250) — matrix operand order repaired and checked by reading, not compiled, not run on hardware
//
#include <hip/hip_runtime.h>
#include <hip/hip_fp16.h>
#include <math.h>

// ---------------------------------------------------------------------------
// CDNA5 (gfx1250) style-transfer net + segment mean.
// Convs = implicit GEMM on v_wmma_f32_16x16x32_f16, register-tiled MTxNT
// (one wave: 16*MT pixels x 16*NT out-channels, MT*NT accumulators).
// Activations NHWC f16  -> A fragments: 2x global_load_b128 per M tile.
// Weights prepacked f16 [tap][co][ciP]; per tap the block cooperatively
// stages its 16*NT-channel slice into LDS (bank-conflict-padded rows);
// B fragments: 2x ds_load_b128 per N tile.
// ---------------------------------------------------------------------------

typedef _Float16 h16;
typedef __attribute__((ext_vector_type(16))) _Float16 v16h;
typedef __attribute__((ext_vector_type(8)))  _Float16 v8h;
typedef __attribute__((ext_vector_type(8)))  float    v8f;

#define TPB 256
#define WAVES_PER_BLOCK 8   // 256 threads = 8 wave32

// ------------------------------------------------- x: NCHW f32 -> NHWC f16
__global__ __launch_bounds__(TPB) void cvt_nhwc_kernel(
    const float* __restrict__ in, h16* __restrict__ out, int C, int HW, int total) {
  int i = blockIdx.x * TPB + threadIdx.x;          // i = NHWC flat index
  if (i >= total) return;
  int ci  = i % C;
  int pix = i / C;
  int n = pix / HW, p = pix - n * HW;
  out[i] = (h16)in[((long)(n * C + ci)) * HW + p];
}

// -------------------------------------- weights -> packed f16 [tap][co][ciP]
// trans=1: torch ConvTranspose2d weight (Cin,Cout,KH,KW), kernel flipped.
__global__ __launch_bounds__(TPB) void prepack_kernel(
    const float* __restrict__ w, h16* __restrict__ wp,
    int Cin, int Cout, int KH, int KW, int CinP, int trans, int total) {
  int i = blockIdx.x * TPB + threadIdx.x;
  if (i >= total) return;
  int ci  = i % CinP;
  int t   = i / CinP;
  int co  = t % Cout;
  int tap = t / Cout;
  int kh = tap / KW, kw = tap - kh * KW;
  float v = 0.f;
  if (ci < Cin) {
    if (trans) v = w[((ci * Cout + co) * KH + (KH - 1 - kh)) * KW + (KW - 1 - kw)];
    else       v = w[((co * Cin + ci) * KH + kh) * KW + kw];
  }
  wp[i] = (h16)v;
}

// ------------------------------------------------------- implicit GEMM conv
// grid: x = pixel-tile blocks (8 waves each), y = out-channel group.
// No early exit: all waves stay alive for the LDS staging barriers; waves
// past M run fully predicated (zero fragments, guarded stores).
template <int KH, int KW, int CINP, int DIL, int REFLECT, int MT, int NT>
__global__ __launch_bounds__(TPB) void conv_wmma_kernel(
    const h16* __restrict__ x, const h16* __restrict__ wpack,
    const float* __restrict__ bias, h16* __restrict__ y16, float* __restrict__ y32,
    int M, int Cin, int Hin, int Win, int Cout, int Hout, int Wout,
    int stride, int pad, int act)
{
  constexpr int KHW  = KH * KW;
  constexpr int NT16 = 16 * NT;
  constexpr int SPITCH = CINP + 8;           // padded LDS row: +16B kills bank conflicts
  const int lane = threadIdx.x & 31;
  const int tmu  = blockIdx.x * WAVES_PER_BLOCK + (threadIdx.x >> 5);
  const int hi = lane >> 4;                  // lane half selects K stripes
  const int lo = lane & 15;
  const int mbase = tmu * (16 * MT);
  const int nbase = blockIdx.y * NT16;       // uniform across the block

  // ---- per-M-tile pixel decode (hoisted; window coords need oy/ox) --------
  int  img[MT], iy0[MT], ix0[MT];
  bool mv[MT];
  #pragma unroll
  for (int i = 0; i < MT; ++i) {
    const int m = mbase + 16 * i + lo;
    mv[i] = (m < M);
    const int mm = mv[i] ? m : 0;
    img[i] = mm / (Hout * Wout);
    const int rp = mm - img[i] * (Hout * Wout);
    const int oy = rp / Wout;
    const int ox = rp - oy * Wout;
    iy0[i] = oy * stride - pad;
    ix0[i] = ox * stride - pad;
  }

  // ---- per-N-tile out-channel decode --------------------------------------
  int co[NT]; bool nv[NT];
  #pragma unroll
  for (int j = 0; j < NT; ++j) {
    const int nch = nbase + 16 * j + lo;
    nv[j] = (nch < Cout);
    co[j] = nv[j] ? nch : 0;                 // clamped; stores are guarded
  }

  const v8f vzero = {};
  v8f acc[MT][NT];
  #pragma unroll
  for (int i = 0; i < MT; ++i)
    #pragma unroll
    for (int j = 0; j < NT; ++j) acc[i][j] = vzero;

  if constexpr (CINP >= 32) {
    // ---------------- fast path: tap loop + LDS-staged B -------------------
    __shared__ h16 smem[NT16 * SPITCH];      // <= ~17 KB (CINP=256, NT=2)
    #pragma unroll
    for (int i = 0; i < MT; ++i) {           // warm WGP$ for the tap reuse
      const int py = iy0[i] < 0 ? 0 : iy0[i];
      const int px = ix0[i] < 0 ? 0 : ix0[i];
      __builtin_prefetch(x + (((long)img[i] * Hin + py) * Win + px) * Cin, 0, 3);
    }

    for (int tap = 0; tap < KHW; ++tap) {
      const int kh = tap / KW, kw = tap - (tap / KW) * KW;   // const divisors

      // cooperative stage of this tap's B slice: [NT16 cols][CINP ci]
      __syncthreads();                       // previous tap's reads complete
      constexpr int VEC = 8;
      for (int t = threadIdx.x; t < (NT16 * CINP) / VEC; t += TPB) {
        const int el  = t * VEC;
        const int col = el / CINP;           // constexpr CINP -> shift
        const int ci  = el - col * CINP;
        const int cog = nbase + col;
        v8h v = {};
        if (cog < Cout)
          v = *(const v8h*)(wpack + ((long)tap * Cout + cog) * CINP + ci);
        *(v8h*)(smem + col * SPITCH + ci) = v;
      }
      __syncthreads();

      // per-wave A addressing for this tap
      const h16* xa[MT]; bool ok[MT];
      #pragma unroll
      for (int i = 0; i < MT; ++i) {
        int iy = iy0[i] + kh, ix = ix0[i] + kw;
        bool o = mv[i];
        if constexpr (REFLECT) {
          iy = iy < 0 ? -iy : (iy >= Hin ? 2 * Hin - 2 - iy : iy);
          ix = ix < 0 ? -ix : (ix >= Win ? 2 * Win - 2 - ix : ix);
        } else if constexpr (DIL == 2) {
          o = o && ((iy | ix) >= 0) && !((iy | ix) & 1);
          iy >>= 1; ix >>= 1;
          o = o && (iy < Hin) && (ix < Win);
        } else {
          o = o && (iy >= 0) && (iy < Hin) && (ix >= 0) && (ix < Win);
        }
        iy = o ? iy : 0;  ix = o ? ix : 0;
        xa[i] = x + (((long)img[i] * Hin + iy) * Win + ix) * Cin;
        ok[i] = o;
      }

      for (int cb = 0; cb < Cin; cb += 32) {
        v16h a[MT], b[NT];
        const v8h z8 = {};
        #pragma unroll
        for (int i = 0; i < MT; ++i) {
          v8h alo = ok[i] ? *(const v8h*)(xa[i] + cb + hi * 8)      : z8;
          v8h ahi = ok[i] ? *(const v8h*)(xa[i] + cb + 16 + hi * 8) : z8;
          a[i] = __builtin_shufflevector(alo, ahi,
                   0,1,2,3,4,5,6,7,8,9,10,11,12,13,14,15);
        }
        #pragma unroll
        for (int j = 0; j < NT; ++j) {
          const h16* srow = smem + (16 * j + lo) * SPITCH + cb + hi * 8;
          v8h blo = *(const v8h*)(srow);       // ds_load_b128
          v8h bhi = *(const v8h*)(srow + 16);  // ds_load_b128
          b[j] = __builtin_shufflevector(blo, bhi,
                   0,1,2,3,4,5,6,7,8,9,10,11,12,13,14,15);
        }
        #pragma unroll
        for (int i = 0; i < MT; ++i)
          #pragma unroll
          for (int j = 0; j < NT; ++j)
            acc[i][j] = __builtin_amdgcn_wmma_f32_16x16x32_f16(
                false, a[i], false, b[j], (short)0, acc[i][j], false, false);
      }
    }
  } else {
    // ---------------- gather path (layer 0, CINP=4, MT=1) ------------------
    constexpr int L2C   = (CINP == 4 ? 2 : (CINP == 8 ? 3 : 4));
    constexpr int KtotP = KHW * CINP;
    constexpr int NCHK  = (KtotP + 31) / 32;
    const long imgHin = (long)img[0] * Hin;
    for (int chk = 0; chk < NCHK; ++chk) {
      const int k0 = chk << 5;
      v16h a;
      #pragma unroll
      for (int e = 0; e < 16; ++e) {
        const int k   = k0 + ((e & 8) << 1) + (hi << 3) + (e & 7);
        const int tap = k >> L2C;
        const int ci  = k & (CINP - 1);
        h16 v = (h16)0;
        if (mv[0] && tap < KHW && ci < Cin) {
          const int kh = tap / KW, kw = tap - (tap / KW) * KW;
          int iy = iy0[0] + kh, ix = ix0[0] + kw;
          if constexpr (REFLECT) {
            iy = iy < 0 ? -iy : (iy >= Hin ? 2 * Hin - 2 - iy : iy);
            ix = ix < 0 ? -ix : (ix >= Win ? 2 * Win - 2 - ix : ix);
          }
          v = x[((imgHin + iy) * Win + ix) * Cin + ci];
        }
        a[e] = v;
      }
      #pragma unroll
      for (int j = 0; j < NT; ++j) {
        v16h b;
        #pragma unroll
        for (int e = 0; e < 16; ++e) {
          const int k   = k0 + ((e & 8) << 1) + (hi << 3) + (e & 7);
          const int tap = k >> L2C;
          const int ci  = k & (CINP - 1);
          h16 v = (h16)0;
          if (tap < KHW)                 // ci >= Cin zero-padded in wpack
            v = wpack[((long)tap * Cout + co[j]) * CINP + ci];
          b[e] = v;
        }
        acc[0][j] = __builtin_amdgcn_wmma_f32_16x16x32_f16(
            false, a, false, b, (short)0, acc[0][j], false, false);
      }
    }
  }

  float bs[NT];
  #pragma unroll
  for (int j = 0; j < NT; ++j) bs[j] = nv[j] ? bias[co[j]] : 0.f;

  // D layout: VGPR r -> row (mtile*16 + r + hi*8), col = lo.
  // NHWC: output pixel flat index == mo, so address is just mo*Cout + ch.
  #pragma unroll
  for (int i = 0; i < MT; ++i) {
    #pragma unroll
    for (int r = 0; r < 8; ++r) {
      const int mo = mbase + 16 * i + r + (hi << 3);
      if (mo < M) {
        const long pbase = (long)mo * Cout;
        #pragma unroll
        for (int j = 0; j < NT; ++j) {
          if (nv[j]) {
            const float v = acc[i][j][r] + bs[j];
            if (act == 1) y32[pbase + nbase + 16 * j + lo] = tanhf(v);
            else          y16[pbase + nbase + 16 * j + lo] = (h16)v;
          }
        }
      }
    }
  }
}

// ------------------------------------ instance norm + relu, NHWC, in place
__global__ __launch_bounds__(TPB) void inorm_relu_kernel(
    h16* __restrict__ y, int C, int plane) {
  const int cch = blockIdx.x % C;
  const int n   = blockIdx.x / C;
  const long base = (long)n * plane * C + cch;
  float s = 0.f, q = 0.f;
  for (int i = threadIdx.x; i < plane; i += TPB) {
    const float v = (float)y[base + (long)i * C];
    s += v; q += v * v;
  }
  __shared__ float ss[TPB], qq[TPB];
  ss[threadIdx.x] = s; qq[threadIdx.x] = q;
  __syncthreads();
  for (int st = TPB / 2; st > 0; st >>= 1) {
    if (threadIdx.x < st) {
      ss[threadIdx.x] += ss[threadIdx.x + st];
      qq[threadIdx.x] += qq[threadIdx.x + st];
    }
    __syncthreads();
  }
  const float inv  = 1.f / (float)plane;
  const float mean = ss[0] * inv;
  const float var  = qq[0] * inv - mean * mean;
  const float rs   = rsqrtf(var + 1e-5f);
  for (int i = threadIdx.x; i < plane; i += TPB) {
    const float v = ((float)y[base + (long)i * C] - mean) * rs;
    y[base + (long)i * C] = (h16)fmaxf(v, 0.f);
  }
}

// --------------------------------------------------------- segment mean
// seg[0..95] = per-(instance,channel) sums, seg[96..127] = counts.
__global__ void seg_zero_kernel(float* __restrict__ seg) {
  if (threadIdx.x < 128) seg[threadIdx.x] = 0.f;
}

__global__ __launch_bounds__(TPB) void seg_accum_kernel(
    const float* __restrict__ y /* NHWC, C=3 */, const int* __restrict__ inst,
    float* __restrict__ seg, int NP) {
  const int i = blockIdx.x * TPB + threadIdx.x;
  if (i >= NP) return;
  int id = inst[i];
  id = id < 0 ? 0 : (id > 31 ? 31 : id);
  atomicAdd(&seg[96 + id], 1.f);
  #pragma unroll
  for (int cc = 0; cc < 3; ++cc)
    atomicAdd(&seg[id * 3 + cc], y[(long)i * 3 + cc]);
}

__global__ __launch_bounds__(TPB) void seg_scatter_kernel(
    const int* __restrict__ inst, const float* __restrict__ seg,
    float* __restrict__ out /* NCHW */, int P, int NP) {
  const int i = blockIdx.x * TPB + threadIdx.x;
  if (i >= NP) return;
  const int n = i / P, p = i - n * P;
  int id = inst[i];
  id = id < 0 ? 0 : (id > 31 ? 31 : id);
  const float cnt = fmaxf(seg[96 + id], 1.f);
  #pragma unroll
  for (int cc = 0; cc < 3; ++cc)
    out[((long)(n * 3 + cc)) * P + p] = seg[id * 3 + cc] / cnt;
}

// ---------------------------------------------------------------- host side
struct ConvCfg {
  int Cin, Hin, Win, Cout, Hout, Wout, stride, pad;
};

template <int KH, int KW, int CINP, int DIL, int REFLECT, int MT, int NT>
static void launch_conv(const h16* x, const h16* wp, const float* b,
                        h16* y16, float* y32, const ConvCfg& g, int act,
                        hipStream_t s) {
  const int Nimg = 8;
  const int M = Nimg * g.Hout * g.Wout;
  const int tiles_mu = (M + 16 * MT - 1) / (16 * MT);
  const int groups   = (g.Cout + 16 * NT - 1) / (16 * NT);
  dim3 grid((tiles_mu + WAVES_PER_BLOCK - 1) / WAVES_PER_BLOCK, groups);
  conv_wmma_kernel<KH, KW, CINP, DIL, REFLECT, MT, NT><<<grid, TPB, 0, s>>>(
      x, wp, b, y16, y32, M, g.Cin, g.Hin, g.Win, g.Cout, g.Hout, g.Wout,
      g.stride, g.pad, act);
}

extern "C" void kernel_launch(void* const* d_in, const int* in_sizes, int n_in,
                              void* d_out, int out_size, void* d_ws, size_t ws_size,
                              hipStream_t stream) {
  (void)in_sizes; (void)n_in; (void)out_size; (void)ws_size;

  const float* x    = (const float*)d_in[0];
  const int*   inst = (const int*)d_in[1];
  const float* W[8]; const float* B[8];
  for (int i = 0; i < 8; ++i) {
    W[i] = (const float*)d_in[2 + 2 * i];
    B[i] = (const float*)d_in[3 + 2 * i];
  }
  float* out = (float*)d_out;

  // ---- scratch layout -----------------------------------------------------
  char* ws = (char*)d_ws;
  const size_t BUFB = (size_t)34u << 20;                 // 2x f16 act buffers
  h16*   bufA = (h16*)ws;
  h16*   bufB = (h16*)(ws + BUFB);
  float* y7   = (float*)(ws + 2 * BUFB);                 // NHWC f32, < 6 MB
  float* seg  = (float*)(ws + 2 * BUFB + ((size_t)6u << 20));
  h16*   wp0  = (h16*)(ws + 2 * BUFB + ((size_t)6u << 20) + 4096);

  // packed weight sub-buffers: [tap][co][ciP], 128-elem aligned
  const int wpElems[8] = {49*32*4, 9*64*32, 9*128*64, 9*256*128,
                          9*128*256, 9*64*128, 9*32*64, 49*3*32};
  h16* WP[8]; { size_t off = 0;
    for (int i = 0; i < 8; ++i) { WP[i] = wp0 + off;
      off += ((size_t)wpElems[i] + 127) & ~(size_t)127; } }

  const int Nimg = 8;
  const int P    = 249 * 249;
  const int NP   = Nimg * P;

  // ---- one-time-per-call prep: weight packing + input transpose -----------
  const int pc[8][6] = { // Cin, Cout, KH, KW, CinP, trans
    {3,32,7,7,4,0},{32,64,3,3,32,0},{64,128,3,3,64,0},{128,256,3,3,128,0},
    {256,128,3,3,256,1},{128,64,3,3,128,1},{64,32,3,3,64,1},{32,3,7,7,32,0}};
  for (int i = 0; i < 8; ++i) {
    const int total = pc[i][2] * pc[i][3] * pc[i][1] * pc[i][4];
    prepack_kernel<<<(total + TPB - 1) / TPB, TPB, 0, stream>>>(
        W[i], WP[i], pc[i][0], pc[i][1], pc[i][2], pc[i][3], pc[i][4],
        pc[i][5], total);
  }
  {
    const int total = 8 * 3 * 256 * 256;
    cvt_nhwc_kernel<<<(total + TPB - 1) / TPB, TPB, 0, stream>>>(
        x, bufB, 3, 256 * 256, total);
  }

  // ---- network ------------------------------------------------------------
  ConvCfg g;

  g = {3,256,256, 32,256,256, 1,3};                       // L0 7x7 reflect
  launch_conv<7,7,4,1,1,1,2>(bufB, WP[0], B[0], bufA, nullptr, g, 0, stream);
  inorm_relu_kernel<<<Nimg * 32, TPB, 0, stream>>>(bufA, 32, 256 * 256);

  g = {32,256,256, 64,128,128, 2,1};                      // L1 down
  launch_conv<3,3,32,1,0,2,2>(bufA, WP[1], B[1], bufB, nullptr, g, 0, stream);
  inorm_relu_kernel<<<Nimg * 64, TPB, 0, stream>>>(bufB, 64, 128 * 128);

  g = {64,128,128, 128,64,64, 2,1};                       // L2 down
  launch_conv<3,3,64,1,0,2,2>(bufB, WP[2], B[2], bufA, nullptr, g, 0, stream);
  inorm_relu_kernel<<<Nimg * 128, TPB, 0, stream>>>(bufA, 128, 64 * 64);

  g = {128,64,64, 256,32,32, 2,1};                        // L3 down
  launch_conv<3,3,128,1,0,2,2>(bufA, WP[3], B[3], bufB, nullptr, g, 0, stream);
  inorm_relu_kernel<<<Nimg * 256, TPB, 0, stream>>>(bufB, 256, 32 * 32);

  g = {256,32,32, 128,63,63, 1,1};                        // L4 up (dilated)
  launch_conv<3,3,256,2,0,2,2>(bufB, WP[4], B[4], bufA, nullptr, g, 0, stream);
  inorm_relu_kernel<<<Nimg * 128, TPB, 0, stream>>>(bufA, 128, 63 * 63);

  g = {128,63,63, 64,125,125, 1,1};                       // L5 up
  launch_conv<3,3,128,2,0,2,2>(bufA, WP[5], B[5], bufB, nullptr, g, 0, stream);
  inorm_relu_kernel<<<Nimg * 64, TPB, 0, stream>>>(bufB, 64, 125 * 125);

  g = {64,125,125, 32,249,249, 1,1};                      // L6 up
  launch_conv<3,3,64,2,0,2,2>(bufB, WP[6], B[6], bufA, nullptr, g, 0, stream);
  inorm_relu_kernel<<<Nimg * 32, TPB, 0, stream>>>(bufA, 32, 249 * 249);

  g = {32,249,249, 3,249,249, 1,3};                       // L7 7x7 reflect+tanh
  launch_conv<7,7,32,1,1,2,1>(bufA, WP[7], B[7], nullptr, y7, g, 1, stream);

  // ---- segment mean -------------------------------------------------------
  seg_zero_kernel<<<1, 128, 0, stream>>>(seg);
  seg_accum_kernel<<<(NP + TPB - 1) / TPB, TPB, 0, stream>>>(y7, inst, seg, NP);
  seg_scatter_kernel<<<(NP + TPB - 1) / TPB, TPB, 0, stream>>>(inst, seg, out, P, NP);
}
